// GemNetT_15281493639548
// MI455X (gfx1250) — matrix operand order repaired and checked by reading, not compiled, not run on hardware
//
#include <hip/hip_runtime.h>
#include <math.h>

// ---------------- problem dims (fixed by reference) ----------------
#define Nc    10000
#define EGc   120000
#define Tc    600000
#define NRc   16
#define NSc   7
#define EAc   128
#define EEc   256
#define NBc   2
#define ECBFc 16
#define INVs  0.70710678118654752440f

typedef _Float16 v16h __attribute__((ext_vector_type(16)));
typedef float    v8f  __attribute__((ext_vector_type(8)));

__device__ __forceinline__ float sc_silu(float x) {
  return x / (1.0f + __expf(-x)) * (1.0f / 0.6f);
}

// ---------------- generic WMMA GEMM ----------------
// C[M,N] = epilogue(A' @ B'); A'/B' built on the fly during f32->f16 LDS staging.
// Pipelined: prefetch tile k+1 into registers while WMMAs consume tile k from LDS.
#define BM 128
#define BN 64
#define BK 32
#define APAD 40   // LDS row stride in halves (bank-conflict padding)

enum { A_PLAIN = 0, A_CONCAT3 = 1, A_BILIN = 2 };
enum { B_PLAIN = 0, B_CBF = 1, B_BILINB = 2 };
enum { E_NONE = 0, E_SILU = 1, E_RES = 2, E_SCATTER = 3 };

struct GemmArgs {
  const float* A;     // PLAIN: MxK (lda). BILIN: x_trip (EG x 64)
  const float* A2;    // CONCAT3: h (indexed by idxA2)
  const float* A3;    // CONCAT3: third piece (per-row, width a3ld)
  const int*   idxA;  // CONCAT3: idx_s ; BILIN: id3_ba
  const int*   idxA2; // CONCAT3: idx_t
  const float* cbf;   // BILIN: T x 16
  const float* B;     // weights (f32)
  const float* res;   // E_RES source
  float*       C;
  const int*   idxC;  // E_SCATTER row map
  int M, N, K, lda, ldc, a3ld;
  float scale;
};

union Frag  { v16h h; uint32_t u[8]; };
union Pack4 { _Float16 h[4]; unsigned long long u; };

// A-tile loader: 8 threads/row * 4 contiguous cols; 4 rows per thread.
template <int AS, bool GUARD>
__device__ __forceinline__ void load_a(const GemmArgs& g, int m0, int k0,
                                       int rA, int cA, float4 (&av)[4]) {
  #pragma unroll
  for (int rr = 0; rr < 4; ++rr) {
    const int r = rr * 32 + rA;
    const int row = m0 + r;
    const int c = k0 + cA;
    float4 t = {0.f, 0.f, 0.f, 0.f};
    if (!GUARD || (row < g.M && c < g.K)) {
      const float* p;
      if (AS == A_PLAIN) {
        p = g.A + row * g.lda + c;
      } else if (AS == A_CONCAT3) {
        if (c < EAc)          p = g.A + g.idxA[row] * EAc + c;
        else if (c < 2 * EAc) p = g.A2 + g.idxA2[row] * EAc + (c - EAc);
        else                  p = g.A3 + row * g.a3ld + (c - 2 * EAc);
      } else {                // A_BILIN: cbf[t, c>>6] * x_trip[id3_ba[t], c&63]
        p = g.A + g.idxA[row] * 64 + (c & 63);
      }
      t = *reinterpret_cast<const float4*>(p);
      if (AS == A_BILIN) {
        const float s = g.cbf[row * 16 + (c >> 6)];
        t.x *= s; t.y *= s; t.z *= s; t.w *= s;
      }
    }
    av[rr] = t;
  }
}

// B-tile loader: chunk = 4 contiguous n at one k; 2 chunks per thread.
template <int BS, bool GUARD>
__device__ __forceinline__ void load_b(const GemmArgs& g, int n0, int k0,
                                       int tid, float4 (&bv)[2]) {
  #pragma unroll
  for (int pass = 0; pass < 2; ++pass) {
    const int id = pass * 256 + tid;
    const int k  = id & 31;
    const int nq = (id >> 5) << 2;
    const int kr = k0 + k, nc = n0 + nq;
    float4 t = {0.f, 0.f, 0.f, 0.f};
    if (!GUARD || (kr < g.K && nc < g.N)) {
      const float* p;
      if (BS == B_PLAIN) {
        p = g.B + kr * g.N + nc;
      } else if (BS == B_CBF) {         // B[r, s*16+i] = W_cbf_down[s][r][i]
        const int s = nc >> 4, i = nc & 15;
        p = g.B + (s * NRc + kr) * ECBFc + i;
      } else {                          // B_BILINB: B[k,o] = Wb[k&63][k>>6][o]
        p = g.B + ((kr & 63) * 16 + (kr >> 6)) * 64 + nc;
      }
      t = *reinterpret_cast<const float4*>(p);
    }
    bv[pass] = t;
  }
}

__device__ __forceinline__ void store_a(_Float16* As, int rA, int cA,
                                        const float4 (&av)[4]) {
  #pragma unroll
  for (int rr = 0; rr < 4; ++rr) {
    Pack4 pk;
    pk.h[0] = (_Float16)av[rr].x; pk.h[1] = (_Float16)av[rr].y;
    pk.h[2] = (_Float16)av[rr].z; pk.h[3] = (_Float16)av[rr].w;
    *reinterpret_cast<unsigned long long*>(&As[(rr * 32 + rA) * APAD + cA]) = pk.u;
  }
}

__device__ __forceinline__ void store_b(_Float16* Bs, int tid, const float4 (&bv)[2]) {
  #pragma unroll
  for (int pass = 0; pass < 2; ++pass) {
    const int id = pass * 256 + tid;
    const int k  = id & 31;
    const int nq = (id >> 5) << 2;
    Bs[(nq + 0) * APAD + k] = (_Float16)bv[pass].x;
    Bs[(nq + 1) * APAD + k] = (_Float16)bv[pass].y;
    Bs[(nq + 2) * APAD + k] = (_Float16)bv[pass].z;
    Bs[(nq + 3) * APAD + k] = (_Float16)bv[pass].w;
  }
}

template <int AS, int BS, int EP>
__global__ __launch_bounds__(256) void gemm_kernel(GemmArgs g) {
  __shared__ _Float16 As[BM * APAD];
  __shared__ _Float16 Bs[BN * APAD];
  const int tid  = threadIdx.x;
  const int m0   = blockIdx.y * BM;
  const int n0   = blockIdx.x * BN;
  const int wv   = tid >> 5, lane = tid & 31;
  const int wm   = wv >> 1, wn = wv & 1;
  const int hi   = lane >> 4, ln = lane & 15;
  const int rA   = tid >> 3;
  const int cA   = (tid & 7) << 2;

  const int M = g.M, N = g.N, K = g.K;
  const bool fullM = (m0 + BM <= M);
  const bool fullN = (n0 + BN <= N);

  float acc[2][2][8];
  #pragma unroll
  for (int a = 0; a < 2; ++a)
    #pragma unroll
    for (int b = 0; b < 2; ++b)
      #pragma unroll
      for (int r = 0; r < 8; ++r) acc[a][b][r] = 0.0f;

  float4 av[4], bv[2], avn[4], bvn[2];
  // initial tile (k0 = 0)
  if (fullM && BK <= K) load_a<AS, false>(g, m0, 0, rA, cA, av);
  else                  load_a<AS, true >(g, m0, 0, rA, cA, av);
  if (fullN && BK <= K) load_b<BS, false>(g, n0, 0, tid, bv);
  else                  load_b<BS, true >(g, n0, 0, tid, bv);

  for (int k0 = 0; k0 < K; k0 += BK) {
    store_a(As, rA, cA, av);
    store_b(Bs, tid, bv);
    __syncthreads();

    // prefetch next tile into registers while WMMAs run on this tile
    const int kn = k0 + BK;
    if (kn < K) {
      if (fullM && kn + BK <= K) load_a<AS, false>(g, m0, kn, rA, cA, avn);
      else                       load_a<AS, true >(g, m0, kn, rA, cA, avn);
      if (fullN && kn + BK <= K) load_b<BS, false>(g, n0, kn, tid, bvn);
      else                       load_b<BS, true >(g, n0, kn, tid, bvn);
    }

    // ---- WMMA compute from LDS ----
    Frag af[2], bf[2];
    #pragma unroll
    for (int tm = 0; tm < 2; ++tm) {
      const int m = wm * 32 + tm * 16 + ln;
      const uint32_t* p = reinterpret_cast<const uint32_t*>(&As[m * APAD]);
      // A 16x32 f16 layout: lane<16 -> K 0..7 & 16..23 ; lane>=16 -> +8
      #pragma unroll
      for (int j = 0; j < 4; ++j) af[tm].u[j]     = p[hi * 4 + j];
      #pragma unroll
      for (int j = 0; j < 4; ++j) af[tm].u[4 + j] = p[8 + hi * 4 + j];
    }
    #pragma unroll
    for (int tn = 0; tn < 2; ++tn) {
      const int n = wn * 32 + tn * 16 + ln;
      const uint32_t* p = reinterpret_cast<const uint32_t*>(&Bs[n * APAD]);
      // B 32x16 f16 layout: lane<16 -> K 0..7 & 16..23 ; lane>=16 -> +8
      #pragma unroll
      for (int j = 0; j < 8; ++j) bf[tn].u[j] = p[(j >> 2) * 8 + hi * 4 + (j & 3)];
    }
    #pragma unroll
    for (int tm = 0; tm < 2; ++tm)
      #pragma unroll
      for (int tn = 0; tn < 2; ++tn) {
        v8f c;
        #pragma unroll
        for (int r = 0; r < 8; ++r) c[r] = acc[tm][tn][r];
        c = __builtin_amdgcn_wmma_f32_16x16x32_f16(
            false, af[tm].h, false, bf[tn].h, (short)0, c, false, false);
        #pragma unroll
        for (int r = 0; r < 8; ++r) acc[tm][tn][r] = c[r];
      }
    __syncthreads();

    #pragma unroll
    for (int j = 0; j < 4; ++j) av[j] = avn[j];
    bv[0] = bvn[0]; bv[1] = bvn[1];
  }

  // ---- epilogue: C/D layout -> m_local = r + 8*hi, n_local = ln ----
  for (int tm = 0; tm < 2; ++tm)
    for (int tn = 0; tn < 2; ++tn) {
      const int mb = m0 + wm * 32 + tm * 16 + 8 * hi;
      const int n  = n0 + wn * 32 + tn * 16 + ln;
      if (n >= N) continue;
      #pragma unroll
      for (int r = 0; r < 8; ++r) {
        const int m = mb + r;
        if (m >= M) continue;
        float v = acc[tm][tn][r];
        if (EP == E_SILU) v = sc_silu(v);
        if (EP == E_RES)  v = (g.res[m * g.ldc + n] + sc_silu(v)) * g.scale;
        if (EP == E_SCATTER) {
          atomicAdd(&g.C[g.idxC[m] * g.ldc + n], v);
        } else {
          g.C[m * g.ldc + n] = v;
        }
      }
    }
}

// ---------------- support kernels ----------------
__global__ void rbf_kernel(const float* __restrict__ D, float* __restrict__ rbf, int n) {
  int e = blockIdx.x * 256 + threadIdx.x;
  if (e >= n) return;
  float d  = D[e] * (1.0f / 6.0f);
  float d2 = d * d, d4 = d2 * d2, d5 = d4 * d, d6 = d5 * d, d7 = d6 * d;
  float env = 1.0f - 21.0f * d5 + 35.0f * d6 - 15.0f * d7;
  env = (d < 1.0f) ? env : 0.0f;
  #pragma unroll
  for (int i = 0; i < NRc; ++i) {
    float off = (float)i * (1.0f / 15.0f);
    float t = d - off;
    rbf[e * NRc + i] = env * __expf(-112.5f * t * t);
  }
}

__global__ void cbf_kernel(const float* __restrict__ cosphi, const float* __restrict__ rbfW1,
                           const int* __restrict__ id3ca, float* __restrict__ cbf, int n) {
  int t = blockIdx.x * 256 + threadIdx.x;
  if (t >= n) return;
  const float cl[NSc] = {0.28209479177387814f, 0.4886025119029199f, 0.6307831305050401f,
                         0.7463526651802308f,  0.8462843753216345f, 0.9356025796273889f,
                         1.0171072362820548f};
  float x = cosphi[t];
  float sph[NSc];
  float Pm2 = 1.0f, Pm1 = x;
  sph[0] = cl[0]; sph[1] = cl[1] * x;
  #pragma unroll
  for (int l = 2; l < NSc; ++l) {
    float P = ((2.0f * l - 1.0f) * x * Pm1 - (l - 1.0f) * Pm2) / (float)l;
    sph[l] = cl[l] * P;
    Pm2 = Pm1; Pm1 = P;
  }
  const float* base = rbfW1 + id3ca[t] * (NSc * ECBFc);
  #pragma unroll
  for (int i = 0; i < ECBFc; i += 4) {
    float4 s = {0.f, 0.f, 0.f, 0.f};
    #pragma unroll
    for (int l = 0; l < NSc; ++l) {
      const float4 w = *reinterpret_cast<const float4*>(base + l * ECBFc + i);
      s.x += sph[l] * w.x; s.y += sph[l] * w.y;
      s.z += sph[l] * w.z; s.w += sph[l] * w.w;
    }
    *reinterpret_cast<float4*>(cbf + t * ECBFc + i) = s;
  }
}

__global__ void ln_kernel(const float* __restrict__ X, const float* __restrict__ w,
                          const float* __restrict__ b, float* __restrict__ out,
                          int rows, int F, int add) {
  int wave = threadIdx.x >> 5, lane = threadIdx.x & 31;
  int row = blockIdx.x * 8 + wave;
  if (row >= rows) return;
  const float* xr = X + row * F;
  float s = 0.0f, ss = 0.0f;
  for (int c = lane; c < F; c += 32) { float v = xr[c]; s += v; ss += v * v; }
  #pragma unroll
  for (int o = 16; o > 0; o >>= 1) { s += __shfl_xor(s, o, 32); ss += __shfl_xor(ss, o, 32); }
  float mu = s / (float)F;
  float var = ss / (float)F - mu * mu;
  float rs = rsqrtf(var + 1e-5f);
  float* orow = out + row * F;
  for (int c = lane; c < F; c += 32) {
    float v = (xr[c] - mu) * rs * w[c] + b[c];
    orow[c] = add ? (orow[c] + v) : v;
  }
}

// float4 elementwise kernels (all sizes are multiples of 4)
__global__ void ew_zero4(float4* p, int n4) {
  int i = blockIdx.x * 256 + threadIdx.x;
  if (i < n4) { float4 z = {0.f, 0.f, 0.f, 0.f}; p[i] = z; }
}
__global__ void ew_mul4(float4* a, const float4* __restrict__ b, int n4) {
  int i = blockIdx.x * 256 + threadIdx.x;
  if (i >= n4) return;
  float4 x = a[i], y = b[i];
  x.x *= y.x; x.y *= y.y; x.z *= y.z; x.w *= y.w;
  a[i] = x;
}
__global__ void ew_axpby4(float4* out, const float4* __restrict__ a,
                          const float4* __restrict__ b, float s, int n4) {
  int i = blockIdx.x * 256 + threadIdx.x;
  if (i >= n4) return;
  float4 x = a[i], y = b[i], r;
  r.x = (x.x + y.x) * s; r.y = (x.y + y.y) * s;
  r.z = (x.z + y.z) * s; r.w = (x.w + y.w) * s;
  out[i] = r;
}
// x = (skip + (ca + ac[swap]) * INV) * INV   (F = 256 -> 64 float4/row)
__global__ void comb_x3(float4* __restrict__ x, const float4* __restrict__ skip,
                        const float4* __restrict__ ca, const float4* __restrict__ ac,
                        const int* __restrict__ sw, int n4) {
  int i = blockIdx.x * 256 + threadIdx.x;
  if (i >= n4) return;
  int e = i >> 6, c = i & 63;
  float4 a = ca[i], b = ac[sw[e] * 64 + c], sk = skip[i], r;
  r.x = (sk.x + (a.x + b.x) * INVs) * INVs;
  r.y = (sk.y + (a.y + b.y) * INVs) * INVs;
  r.z = (sk.z + (a.z + b.z) * INVs) * INVs;
  r.w = (sk.w + (a.w + b.w) * INVs) * INVs;
  x[i] = r;
}
// dst[idx[e]] += s1[e]*s2[e]   (F = 256 -> 64 float4/row)
__global__ void scatter_ma(float* __restrict__ dst, const float4* __restrict__ s1,
                           const float4* __restrict__ s2, const int* __restrict__ idx, int n4) {
  int i = blockIdx.x * 256 + threadIdx.x;
  if (i >= n4) return;
  int e = i >> 6, c = (i & 63) << 2;
  float4 a = s1[i], b = s2[i];
  float* d = dst + idx[e] * 256 + c;
  atomicAdd(d + 0, a.x * b.x);
  atomicAdd(d + 1, a.y * b.y);
  atomicAdd(d + 2, a.z * b.z);
  atomicAdd(d + 3, a.w * b.w);
}
__global__ void gather_h(float4* __restrict__ h, const float4* __restrict__ emb,
                         const int* __restrict__ Zp, int n4) {
  int i = blockIdx.x * 256 + threadIdx.x;
  if (i >= n4) return;
  int a = i >> 5, c = i & 31;   // 128 cols -> 32 float4
  h[i] = emb[Zp[a] * 32 + c];
}

// ---------------- host orchestration ----------------
extern "C" void kernel_launch(void* const* d_in, const int* in_sizes, int n_in,
                              void* d_out, int out_size, void* d_ws, size_t ws_size,
                              hipStream_t stream) {
  (void)in_sizes; (void)n_in; (void)out_size; (void)ws_size;
  const float* Din        = (const float*)d_in[0];
  const float* cosphi     = (const float*)d_in[1];
  const float* atom_emb   = (const float*)d_in[2];
  const float* W_rbf3     = (const float*)d_in[3];
  const float* W_rbf_h    = (const float*)d_in[4];
  const float* W_rbf_out  = (const float*)d_in[5];
  const float* W_cbf_down = (const float*)d_in[6];
  const float* W_edge_emb = (const float*)d_in[7];
  const float* ib_dense_ca      = (const float*)d_in[8];
  const float* ib_trip_dense_ba = (const float*)d_in[9];
  const float* ib_trip_mlp_rbf  = (const float*)d_in[10];
  const float* ib_trip_down     = (const float*)d_in[11];
  const float* ib_trip_bilinear = (const float*)d_in[12];
  const float* ib_trip_up_ca    = (const float*)d_in[13];
  const float* ib_trip_up_ac    = (const float*)d_in[14];
  const float* ib_res_before    = (const float*)d_in[15];
  const float* ib_res_after     = (const float*)d_in[16];
  const float* ib_atom_dense_rbf= (const float*)d_in[17];
  const float* ib_atom_dense1   = (const float*)d_in[18];
  const float* ib_atom_res      = (const float*)d_in[19];
  const float* ib_concat_dense  = (const float*)d_in[20];
  const float* ib_res_m         = (const float*)d_in[21];
  const float* ob_dense_rbf     = (const float*)d_in[22];
  const float* ob_dense1        = (const float*)d_in[23];
  const float* ob_res           = (const float*)d_in[24];
  const float* ln_m_w = (const float*)d_in[25];
  const float* ln_m_b = (const float*)d_in[26];
  const float* ln_E_w = (const float*)d_in[27];
  const float* ln_E_b = (const float*)d_in[28];
  const int* Z       = (const int*)d_in[29];
  const int* idx_s   = (const int*)d_in[30];
  const int* idx_t   = (const int*)d_in[31];
  const int* id3_ba  = (const int*)d_in[32];
  const int* id3_ca  = (const int*)d_in[33];
  const int* id_swap = (const int*)d_in[34];
  float* out = (float*)d_out;

  // workspace layout
  float* ws = (float*)d_ws;
  size_t o = 0;
  auto alloc = [&](size_t nf) { float* p = ws + o; o += nf; return p; };
  float* rbf   = alloc((size_t)EGc * 16);
  float* rbf3  = alloc((size_t)EGc * 16);
  float* rbfh  = alloc((size_t)EGc * 16);
  float* rbfo  = alloc((size_t)EGc * 16);
  float* rbfW1 = alloc((size_t)EGc * 112);
  float* cbf   = alloc((size_t)Tc * 16);
  float* h     = alloc((size_t)Nc * 128);
  float* m     = alloc((size_t)EGc * 256);
  float* x     = alloc((size_t)EGc * 256);
  float* skip  = alloc((size_t)EGc * 256);
  float* tA    = alloc((size_t)EGc * 256);
  float* tB    = alloc((size_t)EGc * 256);
  float* xtrip = alloc((size_t)EGc * 64);
  float* xedge = alloc((size_t)EGc * 64);
  float* xaN   = alloc((size_t)Nc * 256);
  float* nA    = alloc((size_t)Nc * 128);
  float* nB    = alloc((size_t)Nc * 128);

  auto b4 = [](int n4) { return dim3((unsigned)((n4 + 255) / 256)); };

  auto g_silu = [&](const float* A, int lda, const float* W, float* C, int M, int Nn, int K) {
    GemmArgs g{}; g.A = A; g.lda = lda; g.B = W; g.C = C; g.ldc = Nn;
    g.M = M; g.N = Nn; g.K = K;
    dim3 gr((Nn + BN - 1) / BN, (M + BM - 1) / BM);
    gemm_kernel<A_PLAIN, B_PLAIN, E_SILU><<<gr, 256, 0, stream>>>(g);
  };
  auto g_none = [&](const float* A, int lda, const float* W, float* C, int M, int Nn, int K) {
    GemmArgs g{}; g.A = A; g.lda = lda; g.B = W; g.C = C; g.ldc = Nn;
    g.M = M; g.N = Nn; g.K = K;
    dim3 gr((Nn + BN - 1) / BN, (M + BM - 1) / BM);
    gemm_kernel<A_PLAIN, B_PLAIN, E_NONE><<<gr, 256, 0, stream>>>(g);
  };
  auto g_res = [&](const float* A, int lda, const float* W, const float* res, float* C,
                   int M, int Nn, int K) {
    GemmArgs g{}; g.A = A; g.lda = lda; g.B = W; g.res = res; g.C = C; g.ldc = Nn;
    g.M = M; g.N = Nn; g.K = K; g.scale = INVs;
    dim3 gr((Nn + BN - 1) / BN, (M + BM - 1) / BM);
    gemm_kernel<A_PLAIN, B_PLAIN, E_RES><<<gr, 256, 0, stream>>>(g);
  };
  // residual layer: x = (x + silu(silu(x@W0)@W1)) * inv_sqrt2
  auto residual = [&](float* xb, const float* W0, const float* W1, float* tmp, int M, int F) {
    g_silu(xb, F, W0, tmp, M, F, F);
    g_res(tmp, F, W1, xb, xb, M, F, F);
  };
  auto g_concat3 = [&](const float* third, int a3ld, int K, const float* W, float* C) {
    GemmArgs g{}; g.A = h; g.A2 = h; g.A3 = third; g.a3ld = a3ld;
    g.idxA = idx_s; g.idxA2 = idx_t;
    g.B = W; g.C = C; g.ldc = 256; g.M = EGc; g.N = 256; g.K = K;
    dim3 gr(256 / BN, (EGc + BM - 1) / BM);
    gemm_kernel<A_CONCAT3, B_PLAIN, E_SILU><<<gr, 256, 0, stream>>>(g);
  };

  auto out_block = [&](int i, int add) {
    ln_kernel<<<(EGc + 7) / 8, 256, 0, stream>>>(m, ln_m_w + i * 256,
                                                 ln_m_b + i * 256, tA, EGc, 256, 0);
    g_none(rbfo, 16, ob_dense_rbf + (size_t)i * 16 * 256, tB, EGc, 256, 16);
    ew_zero4<<<b4(Nc * 64), 256, 0, stream>>>((float4*)xaN, Nc * 64);
    scatter_ma<<<b4(EGc * 64), 256, 0, stream>>>(xaN, (const float4*)tA, (const float4*)tB,
                                                 idx_t, EGc * 64);
    g_silu(xaN, 256, ob_dense1 + (size_t)i * 256 * 128, nA, Nc, 128, 256);
    for (int r = 0; r < 2; ++r) {
      const float* W0 = ob_res + (((size_t)i * 2 + r) * 2 + 0) * 128 * 128;
      const float* W1 = ob_res + (((size_t)i * 2 + r) * 2 + 1) * 128 * 128;
      residual(nA, W0, W1, nB, Nc, 128);
    }
    ln_kernel<<<(Nc + 7) / 8, 256, 0, stream>>>(nA, ln_E_w + i * 128,
                                                ln_E_b + i * 128, out, Nc, 128, add);
  };

  // ---------------- pipeline ----------------
  rbf_kernel<<<b4(EGc), 256, 0, stream>>>(Din, rbf, EGc);
  g_none(rbf, 16, W_rbf3, rbf3, EGc, 16, 16);
  g_none(rbf, 16, W_rbf_h, rbfh, EGc, 16, 16);
  g_none(rbf, 16, W_rbf_out, rbfo, EGc, 16, 16);
  { // rbf_W1 = einsum('er,sri->esi') as (EG,16)@(16,112) with remapped B
    GemmArgs g{}; g.A = rbf; g.lda = 16; g.B = W_cbf_down; g.C = rbfW1; g.ldc = 112;
    g.M = EGc; g.N = 112; g.K = 16;
    dim3 gr((112 + BN - 1) / BN, (EGc + BM - 1) / BM);
    gemm_kernel<A_PLAIN, B_CBF, E_NONE><<<gr, 256, 0, stream>>>(g);
  }
  cbf_kernel<<<b4(Tc), 256, 0, stream>>>(cosphi, rbfW1, id3_ca, cbf, Tc);
  gather_h<<<b4(Nc * 32), 256, 0, stream>>>((float4*)h, (const float4*)atom_emb, Z, Nc * 32);
  g_concat3(rbf, 16, 2 * EAc + NRc, W_edge_emb, m);   // m = silu(concat@W)

  out_block(0, 0);

  for (int i = 0; i < NBc; ++i) {
    const size_t oEE = (size_t)i * 256 * 256;
    // ---- triplet interaction ----
    g_silu(m, 256, ib_dense_ca + oEE, skip, EGc, 256, 256);
    g_silu(m, 256, ib_trip_dense_ba + oEE, tA, EGc, 256, 256);
    g_none(rbf3, 16, ib_trip_mlp_rbf + (size_t)i * 16 * 256, tB, EGc, 256, 16);
    ew_mul4<<<b4(EGc * 64), 256, 0, stream>>>((float4*)tA, (const float4*)tB, EGc * 64);
    g_silu(tA, 256, ib_trip_down + (size_t)i * 256 * 64, xtrip, EGc, 64, 256);
    ew_zero4<<<b4(EGc * 16), 256, 0, stream>>>((float4*)xedge, EGc * 16);
    { // bilinear as (T,1024)@(1024,64) GEMM + atomic scatter T->EG
      GemmArgs g{}; g.A = xtrip; g.idxA = id3_ba; g.cbf = cbf;
      g.B = ib_trip_bilinear + (size_t)i * 64 * 16 * 64;
      g.C = xedge; g.idxC = id3_ca; g.ldc = 64;
      g.M = Tc; g.N = 64; g.K = 1024;
      dim3 gr(1, (Tc + BM - 1) / BM);
      gemm_kernel<A_BILIN, B_BILINB, E_SCATTER><<<gr, 256, 0, stream>>>(g);
    }
    g_silu(xedge, 64, ib_trip_up_ca + (size_t)i * 64 * 256, tA, EGc, 256, 64);
    g_silu(xedge, 64, ib_trip_up_ac + (size_t)i * 64 * 256, tB, EGc, 256, 64);
    comb_x3<<<b4(EGc * 64), 256, 0, stream>>>((float4*)x, (const float4*)skip,
                                              (const float4*)tA, (const float4*)tB,
                                              id_swap, EGc * 64);
    // ---- edge update ----
    residual(x, ib_res_before + i * 2 * 65536, ib_res_before + i * 2 * 65536 + 65536,
             tA, EGc, 256);
    ew_axpby4<<<b4(EGc * 64), 256, 0, stream>>>((float4*)x, (const float4*)x,
                                                (const float4*)m, INVs, EGc * 64);
    residual(x, ib_res_after + i * 2 * 65536, ib_res_after + i * 2 * 65536 + 65536,
             tA, EGc, 256);
    // ---- atom update ----
    g_none(rbfh, 16, ib_atom_dense_rbf + (size_t)i * 16 * 256, tB, EGc, 256, 16);
    ew_zero4<<<b4(Nc * 64), 256, 0, stream>>>((float4*)xaN, Nc * 64);
    scatter_ma<<<b4(EGc * 64), 256, 0, stream>>>(xaN, (const float4*)x, (const float4*)tB,
                                                 idx_t, EGc * 64);
    g_silu(xaN, 256, ib_atom_dense1 + (size_t)i * 256 * 128, nA, Nc, 128, 256);
    for (int r = 0; r < 2; ++r) {
      const float* W0 = ib_atom_res + (((size_t)i * 2 + r) * 2 + 0) * 128 * 128;
      const float* W1 = ib_atom_res + (((size_t)i * 2 + r) * 2 + 1) * 128 * 128;
      residual(nA, W0, W1, nB, Nc, 128);
    }
    ew_axpby4<<<b4(Nc * 32), 256, 0, stream>>>((float4*)h, (const float4*)h,
                                               (const float4*)nA, INVs, Nc * 32);
    // ---- edge embedding refresh ----
    g_concat3(x, 256, 2 * EAc + EEc, ib_concat_dense + (size_t)i * 512 * 256, tA);
    residual(tA, ib_res_m + i * 2 * 65536, ib_res_m + i * 2 * 65536 + 65536, tB, EGc, 256);
    ew_axpby4<<<b4(EGc * 64), 256, 0, stream>>>((float4*)m, (const float4*)m,
                                                (const float4*)tA, INVs, EGc * 64);
    out_block(i + 1, 1);
  }
}
